// Attention_6296422056433
// MI455X (gfx1250) — matrix operand order repaired and checked by reading, not compiled, not run on hardware
//
#include <hip/hip_runtime.h>
#include <hip/hip_bf16.h>

// MI455X / gfx1250, wave32. Heavy math on v_wmma_f32_16x16x32_bf16.

typedef __attribute__((ext_vector_type(16))) __bf16 v16bf;
typedef __attribute__((ext_vector_type(8)))  float  v8f;
typedef __attribute__((__vector_size__(4 * sizeof(int)))) int v4i;

#define DIM   1024
#define NHEAD 16
#define HD    64
#define BATCH 2
#define SEQ   2048
#define MTOT  (BATCH*SEQ)
#define NKB   (SEQ/64)

// ---- gfx1250 async global->LDS copy path ------------------------------------
#if __has_builtin(__builtin_amdgcn_global_load_async_to_lds_b128) && \
    __has_builtin(__builtin_amdgcn_s_wait_asynccnt)
#define HAVE_ASYNC_LDS 1
#else
#define HAVE_ASYNC_LDS 0
#endif

__device__ __forceinline__ void copy16B_g2l(void* lds, const void* g) {
#if HAVE_ASYNC_LDS
  __builtin_amdgcn_global_load_async_to_lds_b128(
      (__attribute__((address_space(1))) v4i*)g,
      (__attribute__((address_space(3))) v4i*)lds, 0, 0);
#else
  *(uint4*)lds = *(const uint4*)g;
#endif
}

// ---------------- WMMA fragment helpers (wave32 layouts, ISA 7.12.2) ---------

__device__ __forceinline__ v16bf ld_afrag(const __bf16* base, int ld) {
  int t = threadIdx.x & 31;
  int row = t & 15, hi = (t >> 4) & 1;
  const __bf16* p0 = base + row*ld + hi*16;
  const __bf16* p1 = base + row*ld + hi*16 + 8;
  v16bf a;
#pragma unroll
  for (int i = 0; i < 8; ++i) a[i]   = p0[i];
#pragma unroll
  for (int i = 0; i < 8; ++i) a[i+8] = p1[i];
  return a;
}

__device__ __forceinline__ v16bf ld_bfragT(const __bf16* base, int ld) {
  int t = threadIdx.x & 31;
  int col = t & 15, hi = (t >> 4) & 1;
  const __bf16* p = base + col*ld + hi*16;
  v16bf b;
#pragma unroll
  for (int i = 0; i < 16; ++i) b[i] = p[i];
  return b;
}

__device__ __forceinline__ v16bf ld_bfrag_lin(const __bf16* base) {
  int t = threadIdx.x & 31;
  const __bf16* p = base + t*16;
  v16bf b;
#pragma unroll
  for (int i = 0; i < 16; ++i) b[i] = p[i];
  return b;
}

__device__ __forceinline__ void st_dfrag(float* base, int ld, v8f d) {
  int t = threadIdx.x & 31;
  int col = t & 15, rb = ((t >> 4) & 1) * 8;
#pragma unroll
  for (int r = 0; r < 8; ++r) base[(rb+r)*ld + col] = d[r];
}

__device__ __forceinline__ v8f wmma_bf16(v16bf a, v16bf b, v8f c) {
  return __builtin_amdgcn_wmma_f32_16x16x32_bf16(false, a, false, b, (short)0, c,
                                                 false, false);
}

// Fragment-linear index for element (k = kg*16 + i, n) of a 64x64 B tile.
__device__ __forceinline__ int bf_idx(int kg, int n) {
  return (((kg >> 1) * 4 + (n >> 4)) * 32 + ((kg & 1) * 16 + (n & 15))) * 16;
}

// ---------------- Projection / output GEMM (128x64 block, pipelined) ---------
// MODE: 0 = V proj (bf16 head-major out), 1 = K proj (+RoPE),
//       2 = Q proj (+RoPE, *0.125), 3 = final proj (f32 row-major out, bf16 A)
template<int MODE, bool ABF>
__global__ __launch_bounds__(256)
void gemm64(const void* __restrict__ Ap, const float* __restrict__ W,
            const float* __restrict__ cosp, const float* __restrict__ sinp,
            void* __restrict__ outp)
{
  __shared__ __align__(16) __bf16 As[128][72];  // 128 x 64 A tile
  __shared__ __align__(16) __bf16 Bf[4096];     // 64k x 64n B tile, frag-linear
  __shared__ float Ct[128][65];

  const int tid  = threadIdx.x;
  const int m0   = blockIdx.y * 128;
  const int n0   = blockIdx.x * 64;
  const int qs   = tid >> 5;          // wave = row strip (x16), 0..7

  v8f acc[4];
#pragma unroll
  for (int s = 0; s < 4; ++s) acc[s] = (v8f){0,0,0,0,0,0,0,0};

  const int sr = tid >> 1;            // A staging row 0..127
  const int sc = (tid & 1) * 32;      // A staging col chunk (32 elems)
  const int bn  = tid & 63;           // B staging: n
  const int bkg = tid >> 6;           // B staging: k group of 16

  float  aregf[32];
  __bf16 aregb[32];
  float  breg[16];

  auto preload = [&](int k0) {
    if (ABF) {
      const __bf16* src = (const __bf16*)Ap + (size_t)(m0+sr)*DIM + k0 + sc;
#pragma unroll
      for (int i = 0; i < 32; ++i) aregb[i] = src[i];
    } else {
      const float* src = (const float*)Ap + (size_t)(m0+sr)*DIM + k0 + sc;
#pragma unroll
      for (int i = 0; i < 32; ++i) aregf[i] = src[i];
    }
    const float* srcB = W + (size_t)(k0 + bkg*16)*DIM + n0 + bn;
#pragma unroll
    for (int i = 0; i < 16; ++i) breg[i] = srcB[(size_t)i*DIM];
  };

  preload(0);

  for (int k0 = 0; k0 < DIM; k0 += 64) {
    // ---- commit prefetched tiles to LDS ----
#pragma unroll
    for (int i = 0; i < 32; ++i)
      As[sr][sc+i] = ABF ? aregb[i] : (__bf16)aregf[i];
    {
      __bf16* dst = &Bf[bf_idx(bkg, bn)];
#pragma unroll
      for (int i = 0; i < 16; ++i) dst[i] = (__bf16)breg[i];
    }
    if (k0 + 64 < DIM) preload(k0 + 64);   // overlaps with WMMA phase
    __syncthreads();

#pragma unroll
    for (int ks = 0; ks < 64; ks += 32) {
      const int kstep = ks >> 5;
      v16bf a = ld_afrag(&As[qs*16][ks], 72);
#pragma unroll
      for (int s = 0; s < 4; ++s) {
        v16bf b = ld_bfrag_lin(Bf + (kstep*4 + s) * 512);
        acc[s] = wmma_bf16(a, b, acc[s]);
      }
    }
    __syncthreads();
  }

#pragma unroll
  for (int s = 0; s < 4; ++s)
    st_dfrag(&Ct[qs*16][s*16], 65, acc[s]);
  __syncthreads();

  // ---- epilogue: contiguous 32-element chunk per thread ----
  {
    const int row = sr;
    const int m   = m0 + row;
    if (MODE == 3) {
      float* dst = (float*)outp + (size_t)m*DIM + n0 + sc;
#pragma unroll
      for (int j = 0; j < 32; ++j) dst[j] = Ct[row][sc+j];
    } else {
      const int s    = m & (SEQ-1);
      const int bidx = m >> 11;
      const int h    = n0 >> 6;
      __bf16* dst = (__bf16*)outp + ((size_t)(bidx*NHEAD + h)*SEQ + s)*HD + sc;
#pragma unroll
      for (int j = 0; j < 32; ++j) {
        int col = sc + j;
        float v = Ct[row][col];
        if (MODE >= 1) {
          if (col < 32) {
            float v2 = Ct[row][col+32];
            v = v * cosp[s*32 + col] - v2 * sinp[s*32 + col];
          } else {
            float v2 = Ct[row][col-32];
            v = v * cosp[s*32 + col-32] + v2 * sinp[s*32 + col-32];
          }
        }
        if (MODE == 2) v *= 0.125f;
        dst[j] = (__bf16)v;
      }
    }
  }
}

// -------- Flash attention: 128-query block, register O, pipelined ------------
__global__ __launch_bounds__(256)
void attn64(const __bf16* __restrict__ Q, const __bf16* __restrict__ K,
            const __bf16* __restrict__ V, __bf16* __restrict__ O)
{
  __shared__ __align__(16) __bf16 Qs[128][72];    // q x d
  __shared__ __align__(16) __bf16 Ks[2][64][72];  // double-buffered key x d
  __shared__ __align__(16) __bf16 Vf[4096];       // V tile, fragment-linear
  __shared__ __align__(16) __bf16 Pb[128][72];    // probs q x key
  __shared__ float Sb[128][65];                   // scores / final staging
  __shared__ float mrow[128], lrow[128], arow[128];

  const int tid  = threadIdx.x;
  const int q0   = blockIdx.x * 128;
  const int bh   = blockIdx.y;
  const size_t base = (size_t)bh * SEQ * HD;
  const int qs   = tid >> 5;                      // wave = q strip (x16)
  const int lane = tid & 31;
  const int lrb  = ((lane >> 4) & 1) * 8;         // C-frag row base offset

  const int sr  = tid >> 1;           // 128-row staging: row
  const int sc  = (tid & 1) * 32;     // 128-row staging: col chunk (32)
  const int sr4 = tid >> 2;           // 64-row staging: row
  const int sc4 = (tid & 3) * 16;     // 64-row staging: col chunk (16)
  const int vd  = tid & 63;           // V staging: d
  const int vkg = tid >> 6;           // V staging: key group of 16

  v8f oc[4];
#pragma unroll
  for (int s = 0; s < 4; ++s) oc[s] = (v8f){0,0,0,0,0,0,0,0};
  __bf16 vreg[16];

  auto issueK = [&](int kb, int buf) {
    const size_t rbase = base + (size_t)(kb*64 + sr4)*HD + sc4;
    copy16B_g2l(&Ks[buf][sr4][sc4],   K + rbase);
    copy16B_g2l(&Ks[buf][sr4][sc4+8], K + rbase + 8);
  };
  auto loadV = [&](int kb) {
    const __bf16* src = V + base + (size_t)(kb*64 + vkg*16)*HD + vd;
#pragma unroll
    for (int i = 0; i < 16; ++i) vreg[i] = src[(size_t)i*HD];
  };

  // prologue: Q (4 async/thread) + first K block async, first V into regs
  {
    const size_t qbase = base + (size_t)(q0+sr)*HD + sc;
#pragma unroll
    for (int c = 0; c < 4; ++c)
      copy16B_g2l(&Qs[sr][sc + c*8], Q + qbase + c*8);
  }
  issueK(0, 0);
  loadV(0);
  if (tid < 128) { mrow[tid] = -1e30f; lrow[tid] = 0.f; }
#if HAVE_ASYNC_LDS
  __builtin_amdgcn_s_wait_asynccnt(0);
#endif
  __syncthreads();

  for (int kb = 0; kb < NKB; ++kb) {
    const int cur = kb & 1;
    // commit prefetched V registers to fragment-linear LDS
    {
      __bf16* dst = &Vf[bf_idx(vkg, vd)];
#pragma unroll
      for (int i = 0; i < 16; ++i) dst[i] = vreg[i];
    }
    // prefetch next block (K async into other buffer, V into registers)
    if (kb + 1 < NKB) {
      issueK(kb + 1, cur ^ 1);
      loadV(kb + 1);
#if HAVE_ASYNC_LDS
      __builtin_amdgcn_s_wait_asynccnt(2);   // only next block in flight
#endif
    } else {
#if HAVE_ASYNC_LDS
      __builtin_amdgcn_s_wait_asynccnt(0);
#endif
    }
    __syncthreads();

    // ---- S = Q K^T: 4 key-subtiles per wave ----
    {
      v8f sacc[4];
#pragma unroll
      for (int s = 0; s < 4; ++s) sacc[s] = (v8f){0,0,0,0,0,0,0,0};
#pragma unroll
      for (int ks = 0; ks < HD; ks += 32) {
        v16bf a = ld_afrag(&Qs[qs*16][ks], 72);
#pragma unroll
        for (int s = 0; s < 4; ++s) {
          v16bf b = ld_bfragT(&Ks[cur][s*16][ks], 72);
          sacc[s] = wmma_bf16(a, b, sacc[s]);
        }
      }
#pragma unroll
      for (int s = 0; s < 4; ++s)
        st_dfrag(&Sb[qs*16][s*16], 65, sacc[s]);
    }
    __syncthreads();

    // ---- online softmax: 2 threads per row, width-2 shuffle reduction ----
    {
      const int rrow = tid >> 1;
      const int rseg = (tid & 1) * 32;
      float pm = -1e30f;
#pragma unroll
      for (int j = 0; j < 32; ++j) pm = fmaxf(pm, Sb[rrow][rseg+j]);
      pm = fmaxf(pm, __shfl_xor(pm, 1, 32));
      float mo = mrow[rrow];
      float mx = fmaxf(mo, pm);
      float alpha = __expf(mo - mx);
      float l = 0.f;
#pragma unroll
      for (int j = 0; j < 32; ++j) {
        float p = __expf(Sb[rrow][rseg+j] - mx);
        l += p;
        Pb[rrow][rseg+j] = (__bf16)p;
      }
      l += __shfl_xor(l, 1, 32);
      if ((tid & 1) == 0) {
        mrow[rrow] = mx;
        lrow[rrow] = lrow[rrow] * alpha + l;
        arow[rrow] = alpha;
      }
    }
    __syncthreads();

    // ---- rescale register accumulators, then O += P @ V ----
#pragma unroll
    for (int r = 0; r < 8; ++r) {
      float al = arow[qs*16 + lrb + r];
#pragma unroll
      for (int s = 0; s < 4; ++s) oc[s][r] *= al;
    }
#pragma unroll
    for (int ks = 0; ks < 64; ks += 32) {
      v16bf a = ld_afrag(&Pb[qs*16][ks], 72);
#pragma unroll
      for (int s = 0; s < 4; ++s) {
        v16bf b = ld_bfrag_lin(Vf + ((ks>>5)*4 + s) * 512);
        oc[s] = wmma_bf16(a, b, oc[s]);
      }
    }
    __syncthreads();   // protect Vf/Pb before next iteration's staging
  }

  // ---- epilogue: stage O through Sb, normalize, store ----
#pragma unroll
  for (int s = 0; s < 4; ++s)
    st_dfrag(&Sb[qs*16][s*16], 65, oc[s]);
  __syncthreads();
  {
    const int b = bh >> 4, h = bh & 15;
    float inv = 1.0f / lrow[sr];
    __bf16* dst = O + ((size_t)(b*SEQ + q0 + sr))*DIM + h*HD + sc;
#pragma unroll
    for (int j = 0; j < 32; ++j) dst[j] = (__bf16)(Sb[sr][sc+j] * inv);
  }
}

// ---------------- launch ------------------------------------------------------
extern "C" void kernel_launch(void* const* d_in, const int* in_sizes, int n_in,
                              void* d_out, int out_size, void* d_ws, size_t ws_size,
                              hipStream_t stream)
{
  const float* x    = (const float*)d_in[0];
  const float* cosp = (const float*)d_in[1];
  const float* sinp = (const float*)d_in[2];
  const float* Wq   = (const float*)d_in[3];
  const float* Wk   = (const float*)d_in[4];
  const float* Wv   = (const float*)d_in[5];
  const float* Wo   = (const float*)d_in[6];

  // Workspace: Q,K,V head-major bf16 + attention output bf16 = 32 MiB
  const size_t QKV = (size_t)BATCH * SEQ * DIM;
  __bf16* Qb = (__bf16*)d_ws;
  __bf16* Kb = Qb + QKV;
  __bf16* Vb = Kb + QKV;
  __bf16* Ab = Vb + QKV;

  dim3 blk(256);
  dim3 g(DIM/64, MTOT/128);
  gemm64<2, false><<<g, blk, 0, stream>>>(x, Wq, cosp, sinp, Qb);
  gemm64<1, false><<<g, blk, 0, stream>>>(x, Wk, cosp, sinp, Kb);
  gemm64<0, false><<<g, blk, 0, stream>>>(x, Wv, cosp, sinp, Vb);

  attn64<<<dim3(SEQ/128, BATCH*NHEAD), blk, 0, stream>>>(Qb, Kb, Vb, Ab);

  gemm64<3, true><<<g, blk, 0, stream>>>(Ab, Wo, nullptr, nullptr, (float*)d_out);
}
